// NCNPredictor_541165879726
// MI455X (gfx1250) — matrix operand order repaired and checked by reading, not compiled, not run on hardware
//
#include <hip/hip_runtime.h>
#include <hip/hip_bf16.h>

#define N_NODES   10000
#define N_EDGES   320000
#define N_PAIRS   8192
#define IN_CH     128
#define HIDDEN    256
#define ADJ_W     313            // ceil(10000/32) words of bitset per row
#define ADJ_STRIDE 320           // padded row stride (words) for alignment
#define ADJ_WORDS  (N_NODES * ADJ_STRIDE)       // 3,200,000 words
#define ADJ_BYTES  ((size_t)ADJ_WORDS * 4u)     // 12.8 MB
#define CN_CAP    2048

typedef __attribute__((ext_vector_type(2)))  float    v2f;
typedef __attribute__((ext_vector_type(8)))  float    v8f;
typedef __attribute__((ext_vector_type(16))) _Float16 v16h;

// Feature-test only meaningful in the device pass; host pass takes the f16
// path purely for parsing (codegen-confirmed builtin, known to parse via
// aux-target registration).
#if defined(__AMDGCN__) && __has_builtin(__builtin_amdgcn_wmma_f32_16x16x4_f32)
#define USE_WMMA_F32X4 1
#else
#define USE_WMMA_F32X4 0
#endif

// ---------------- kernel 0: zero the adjacency bitset ----------------
__global__ void adj_zero_kernel(unsigned* __restrict__ p, int n) {
    int i = blockIdx.x * blockDim.x + threadIdx.x;
    int stride = gridDim.x * blockDim.x;
    for (; i < n; i += stride) p[i] = 0u;
}

// ---------------- kernel 1: build symmetric adjacency bitset ----------------
__global__ void adj_build_kernel(const int* __restrict__ ei, unsigned* __restrict__ adj) {
    int e = blockIdx.x * blockDim.x + threadIdx.x;
    if (e >= N_EDGES) return;
    int u = ei[e];
    int v = ei[N_EDGES + e];
    atomicOr(&adj[(size_t)u * ADJ_STRIDE + (v >> 5)], 1u << (v & 31));
    atomicOr(&adj[(size_t)v * ADJ_STRIDE + (u >> 5)], 1u << (u & 31));
}

// ---------------- kernel 2: per-pair xij + common-neighbor embedding ----------------
// One 128-thread block per pair; thread t owns channel t.
__global__ void __launch_bounds__(128)
pair_kernel(const float* __restrict__ x, const int* __restrict__ tar,
            const unsigned* __restrict__ adj, float* __restrict__ xs) {
    __shared__ int s_cnt;
    __shared__ int s_list[CN_CAP];

    const int b = blockIdx.x;
    const int t = threadIdx.x;              // channel 0..127
    const int i = tar[b];
    const int j = tar[N_PAIRS + b];

    if (t == 0) s_cnt = 0;
    __syncthreads();

    // Scan bitset AND of the two rows; push common-neighbor ids into LDS.
    const unsigned* rI = adj + (size_t)i * ADJ_STRIDE;
    const unsigned* rJ = adj + (size_t)j * ADJ_STRIDE;
    for (int k = t; k < ADJ_W; k += 128) {
        unsigned w = rI[k] & rJ[k];
        while (w) {
            int bit = __builtin_ctz(w);
            w &= (w - 1u);
            int pos = atomicAdd(&s_cnt, 1);
            if (pos < CN_CAP) s_list[pos] = (k << 5) + bit;
        }
    }
    __syncthreads();

    int cnt = s_cnt;
    if (cnt > CN_CAP) cnt = CN_CAP;

    float xij = x[(size_t)i * IN_CH + t] * x[(size_t)j * IN_CH + t];
    float acc = 0.0f;
    for (int l = 0; l < cnt; ++l) {
        int n = s_list[l];                  // LDS broadcast
        acc += x[(size_t)n * IN_CH + t];    // coalesced across channels
    }
    xs[(size_t)b * (2 * IN_CH) + t]          = xij;
    xs[(size_t)b * (2 * IN_CH) + IN_CH + t]  = acc;
}

// ---------------- kernel 3: fused MLP via WMMA ----------------
// out[m] = relu(xs[m,:] @ W1^T + b1) @ W2^T + b2
// Block = 256 threads (8 waves), covers a 16-pair tile across all 256 hidden units.
__global__ void __launch_bounds__(256)
mlp_wmma_kernel(const float* __restrict__ xs, const float* __restrict__ W1,
                const float* __restrict__ b1, const float* __restrict__ W2,
                const float* __restrict__ b2, float* __restrict__ out) {
    __shared__ float sXs[16 * HIDDEN];      // 16 KB: xs tile, row-major
    __shared__ float sOut[16];

    const int row0 = blockIdx.x * 16;

    for (int idx = threadIdx.x; idx < 16 * HIDDEN; idx += 256)
        sXs[idx] = xs[(size_t)row0 * HIDDEN + idx];
    if (threadIdx.x < 16) sOut[threadIdx.x] = b2[0];
    __syncthreads();

    const int wave = threadIdx.x >> 5;
    const int lane = threadIdx.x & 31;
    const int half = lane >> 4;
    const int r    = lane & 15;             // row within A / column (N) within B

    for (int tile = 0; tile < 2; ++tile) {
        const int n0 = wave * 32 + tile * 16;        // hidden column base
        v8f c = {};

#if USE_WMMA_F32X4
        // fp32 path: V_WMMA_F32_16X16X4_F32, exact fp32 like the reference.
        // A 16x4 f32: lane(r,half) holds A[r][k0+2*half], A[r][k0+2*half+1].
        // B[k][n] = W1[n0+r][k]  (h = xs @ W1^T) -> same striding as A.
        const float* arow = sXs + r * HIDDEN + 2 * half;
        const float* brow = W1 + (size_t)(n0 + r) * HIDDEN + 2 * half;
        #pragma unroll 4
        for (int k0 = 0; k0 < HIDDEN; k0 += 4) {
            v2f a, bf;
            a.x  = arow[k0];   a.y  = arow[k0 + 1];
            bf.x = brow[k0];   bf.y = brow[k0 + 1];
            c = __builtin_amdgcn_wmma_f32_16x16x4_f32(false, a, false, bf,
                                                      (short)0, c, false, false);
        }
#else
        // f16 path: V_WMMA_F32_16X16X32_F16 (codegen-confirmed), fp32 accumulate.
        // A 16x32 f16 layout (ISA 7.12.2): VGPRs 0-3 cover K = 8*half + 0..7,
        // VGPRs 4-7 cover K = 8*half + 16..23. B 32x16: same K order as A
        // at column N = r.
        const float* arow = sXs + r * HIDDEN;
        const float* brow = W1 + (size_t)(n0 + r) * HIDDEN;
        for (int k0 = 0; k0 < HIDDEN; k0 += 32) {
            v16h a, bf;
            #pragma unroll
            for (int e = 0; e < 16; ++e) {
                int vg = e >> 1;
                int ka = (vg < 4) ? (2 * vg + (e & 1) + 8 * half)
                                  : (16 + 2 * (vg - 4) + (e & 1) + 8 * half);
                a[e]  = (_Float16)arow[k0 + ka];
                bf[e] = (_Float16)brow[k0 + ka];
            }
            c = __builtin_amdgcn_wmma_f32_16x16x32_f16(false, a, false, bf,
                                                       (short)0, c, false, false);
        }
#endif

        // D layout: VGPR v, lanes 0-15 -> (M=v, N=lane); lanes 16-31 -> (M=v+8)
        const float w2n = W2[n0 + r];
        const float b1n = b1[n0 + r];
        #pragma unroll
        for (int v = 0; v < 8; ++v) {
            float h = c[v] + b1n;
            h = h > 0.0f ? h : 0.0f;
            atomicAdd(&sOut[v + 8 * half], h * w2n);   // ds_add_f32 reduction over n
        }
    }
    __syncthreads();
    if (threadIdx.x < 16) out[row0 + threadIdx.x] = sOut[threadIdx.x];
}

// ---------------- launcher ----------------
extern "C" void kernel_launch(void* const* d_in, const int* in_sizes, int n_in,
                              void* d_out, int out_size, void* d_ws, size_t ws_size,
                              hipStream_t stream) {
    const float* x   = (const float*)d_in[0];
    const int*   ei  = (const int*)  d_in[1];   // edge_index [2, N_EDGES]
    const int*   tar = (const int*)  d_in[2];   // tar_ei     [2, N_PAIRS]
    const float* W1  = (const float*)d_in[3];
    const float* b1  = (const float*)d_in[4];
    const float* W2  = (const float*)d_in[5];
    const float* b2  = (const float*)d_in[6];
    float* out = (float*)d_out;

    unsigned* adj = (unsigned*)d_ws;                       // 12.8 MB bitset
    float*    xs  = (float*)((char*)d_ws + ADJ_BYTES);     // 8 MB [8192, 256]

    // 0) zero adjacency bitset
    adj_zero_kernel<<<2048, 256, 0, stream>>>(adj, ADJ_WORDS);
    // 1) build symmetric adjacency
    adj_build_kernel<<<(N_EDGES + 255) / 256, 256, 0, stream>>>(ei, adj);
    // 2) per-pair features
    pair_kernel<<<N_PAIRS, 128, 0, stream>>>(x, tar, adj, xs);
    // 3) fused WMMA MLP
    mlp_wmma_kernel<<<N_PAIRS / 16, 256, 0, stream>>>(xs, W1, b1, W2, b2, out);
}
